// DDPGHedge_48473000902878
// MI455X (gfx1250) — matrix operand order repaired and checked by reading, not compiled, not run on hardware
//
#include <hip/hip_runtime.h>
#include <hip/hip_bf16.h>

typedef __attribute__((ext_vector_type(16))) _Float16 v16h;
typedef __attribute__((ext_vector_type(8)))  float    v8f;
typedef __attribute__((ext_vector_type(4)))  float    v4f;

#define STATE_DIM 16
#define HIDDEN    64
#define NSTEPS    512
#define WAVES_PER_BLOCK 8
#define TILES_PER_WAVE  2

__device__ __forceinline__ float fast_tanh(float x) {
    // tanh(x) = 1 - 2/(e^{2x}+1); exp2-based, saturates to +/-1 correctly.
    float t = __builtin_amdgcn_exp2f(x * 2.8853900817779268f);   // e^{2x}
    return 1.0f - 2.0f * __builtin_amdgcn_rcpf(t + 1.0f);
}

// One wave32 handles TWO independent tiles of 16 batch rows for all 512 steps.
// The two recurrence chains share the weight fragments and interleave to hide
// WMMA latency. Per tile per step:
//   GEMM1: H1^T = W1^T x X^T   (bias b1 folded at K=17, constant 1 injected in X^T)
//   GEMM2: H2   = H1 x W2      (A = GEMM1 C-frags reinterpreted in-lane,
//                               C-init = b2 broadcast)
//   GEMM3: delta = tanh(H2 x W3 + b3) via VALU + shfl_xor butterfly.
__global__ __launch_bounds__(WAVES_PER_BLOCK * 32)
void DDPGHedge_kernel(const float* __restrict__ features,
                      const float* __restrict__ W1,
                      const float* __restrict__ b1,
                      const float* __restrict__ W2,
                      const float* __restrict__ b2,
                      const float* __restrict__ W3,
                      const float* __restrict__ b3p,
                      float* __restrict__ out,
                      int batch)
{
    const int lane  = threadIdx.x & 31;
    const int wave  = threadIdx.x >> 5;
    const int tile0 = (blockIdx.x * WAVES_PER_BLOCK + wave) * TILES_PER_WAVE;
    if (tile0 * 16 >= batch) return;              // wave-uniform; EXEC stays all-ones
    const int col = lane & 15;                    // batch column within a tile
    const int hi  = lane >> 4;                    // half-wave

    // ---------------- static W1^T A-fragments (4 M-tiles over hidden) -------------
    // A layout: row = lane&15 (-> hidden 16*mt+col), elem i -> K = 16*hi+i
    v16h aW1[4];
#pragma unroll
    for (int mt = 0; mt < 4; ++mt) {
        const int h = 16 * mt + col;
#pragma unroll
        for (int i = 0; i < 16; ++i) {
            const int k = 16 * hi + i;
            float v = 0.0f;
            if (k < 17)       v = W1[k * HIDDEN + h];
            else if (k == 17) v = b1[h];          // bias row: X^T carries 1.0 at K=17
            aW1[mt][i] = (_Float16)v;
        }
    }

    // ---------------- static W2 B-fragments [kc][nt] ------------------------------
    // B layout: col = lane&15 (-> hidden-out 16*nt+col), elem j -> K = 32*kc+((j<8)? 8*hi+j : 16+8*hi+(j-8))
    v16h bW2[2][4];
#pragma unroll
    for (int kc = 0; kc < 2; ++kc)
#pragma unroll
        for (int nt = 0; nt < 4; ++nt) {
            const int n = 16 * nt + col;
#pragma unroll
            for (int j = 0; j < 16; ++j) {
                const int k = 32 * kc + ((j < 8) ? (8 * hi + j) : (16 + 8 * hi + (j - 8)));
                bW2[kc][nt][j] = (_Float16)W2[k * HIDDEN + n];
            }
        }

    float b2loc[4], w3loc[4];
#pragma unroll
    for (int nt = 0; nt < 4; ++nt) {
        b2loc[nt] = b2[16 * nt + col];
        w3loc[nt] = W3[16 * nt + col];
    }
    const float b3 = b3p[0];

    const float* fptr[TILES_PER_WAVE];
    float*       optr[TILES_PER_WAVE];
#pragma unroll
    for (int c = 0; c < TILES_PER_WAVE; ++c) {
        const int b0 = (tile0 + c) * 16;
        fptr[c] = features + (size_t)(b0 + col) * NSTEPS * STATE_DIM;
        optr[c] = out      + (size_t)(b0 + col) * NSTEPS;
    }

    float prev_delta[TILES_PER_WAVE];
    v4f   outv[TILES_PER_WAVE];
    v16h  bx[TILES_PER_WAVE];
#pragma unroll
    for (int c = 0; c < TILES_PER_WAVE; ++c) {
        prev_delta[c] = 0.0f;
        outv[c] = v4f{};
#pragma unroll
        for (int j = 0; j < 16; ++j) bx[c][j] = (_Float16)0.0f;
    }

#pragma unroll 4
    for (int t = 0; t < NSTEPS; ++t) {
        // ---- B_x (X^T): lanes 0-15 carry features K=0..15 of batch col;
        //      lanes 16-31: K=16 = prev_delta, K=17 = 1.0, K>=18 stay 0 (loop-carried).
#pragma unroll
        for (int c = 0; c < TILES_PER_WAVE; ++c) {
            if (hi == 0) {
                const float* fp = fptr[c] + t * STATE_DIM;
                v4f f0 = *(const v4f*)(fp + 0);
                v4f f1 = *(const v4f*)(fp + 4);
                v4f f2 = *(const v4f*)(fp + 8);
                v4f f3 = *(const v4f*)(fp + 12);
#pragma unroll
                for (int j = 0; j < 4; ++j) {
                    bx[c][j]      = (_Float16)f0[j];
                    bx[c][4 + j]  = (_Float16)f1[j];
                    bx[c][8 + j]  = (_Float16)f2[j];
                    bx[c][12 + j] = (_Float16)f3[j];
                }
                if (t + 8 < NSTEPS)
                    __builtin_prefetch(fptr[c] + (t + 8) * STATE_DIM, 0, 3);
            } else {
                bx[c][0] = (_Float16)prev_delta[c];   // K = 16
                bx[c][1] = (_Float16)1.0f;            // K = 17 (bias row of aW1)
            }
        }

        // ---- GEMM1: c1 = W1^T[mt] x X^T  (H1^T tiles: M=hidden, N=batch)
        v8f c1[TILES_PER_WAVE][4];
#pragma unroll
        for (int c = 0; c < TILES_PER_WAVE; ++c)
#pragma unroll
            for (int mt = 0; mt < 4; ++mt) {
                v8f z = {};
                c1[c][mt] = __builtin_amdgcn_wmma_f32_16x16x32_f16(
                    false, aW1[mt], false, bx[c], (short)0, z, false, false);
            }

        // ---- relu + in-lane reinterpret of C-frags as A-frags of H1 (batch x 64)
        v16h aH[TILES_PER_WAVE][2];
#pragma unroll
        for (int c = 0; c < TILES_PER_WAVE; ++c)
#pragma unroll
            for (int kc = 0; kc < 2; ++kc)
#pragma unroll
                for (int i = 0; i < 16; ++i) {
                    float v = (i < 8) ? c1[c][2 * kc][i] : c1[c][2 * kc + 1][i - 8];
                    aH[c][kc][i] = (_Float16)fmaxf(v, 0.0f);
                }

        // ---- GEMM2: c2 = H1 x W2 (+ b2 via C-init). M=batch, N=hidden-out.
        v8f c2[TILES_PER_WAVE][4];
#pragma unroll
        for (int c = 0; c < TILES_PER_WAVE; ++c)
#pragma unroll
            for (int nt = 0; nt < 4; ++nt) {
                v8f cinit;
#pragma unroll
                for (int v = 0; v < 8; ++v) cinit[v] = b2loc[nt];
                v8f acc = __builtin_amdgcn_wmma_f32_16x16x32_f16(
                    false, aH[c][0], false, bW2[0][nt], (short)0, cinit, false, false);
                c2[c][nt] = __builtin_amdgcn_wmma_f32_16x16x32_f16(
                    false, aH[c][1], false, bW2[1][nt], (short)0, acc, false, false);
            }

        // ---- GEMM3 (64 -> 1) via VALU: p[v] = sum_nt relu(c2)*w3, then butterfly
        float p[TILES_PER_WAVE][8];
#pragma unroll
        for (int c = 0; c < TILES_PER_WAVE; ++c)
#pragma unroll
            for (int v = 0; v < 8; ++v) {
                float s = 0.0f;
#pragma unroll
                for (int nt = 0; nt < 4; ++nt)
                    s += fmaxf(c2[c][nt][v], 0.0f) * w3loc[nt];
                p[c][v] = s;
            }
#pragma unroll
        for (int m = 1; m < 16; m <<= 1)
#pragma unroll
            for (int c = 0; c < TILES_PER_WAVE; ++c)
#pragma unroll
                for (int v = 0; v < 8; ++v)
                    p[c][v] += __shfl_xor(p[c][v], m, 32);
        // each lane now holds totals for batch rows v + 8*hi of its half

#pragma unroll
        for (int c = 0; c < TILES_PER_WAVE; ++c) {
            // pick this lane's row (r = lane&15): value index r&7, from half r>>3
            const int r7 = lane & 7;
            float s = p[c][0];
            s = (r7 == 1) ? p[c][1] : s;
            s = (r7 == 2) ? p[c][2] : s;
            s = (r7 == 3) ? p[c][3] : s;
            s = (r7 == 4) ? p[c][4] : s;
            s = (r7 == 5) ? p[c][5] : s;
            s = (r7 == 6) ? p[c][6] : s;
            s = (r7 == 7) ? p[c][7] : s;
            float sx  = __shfl_xor(s, 16, 32);
            float raw = (hi == ((lane & 15) >> 3)) ? s : sx;

            float delta   = fast_tanh(raw + b3);
            prev_delta[c] = delta;                 // feeds next step's bx (K=16)

            outv[c][t & 3] = delta;
            if ((t & 3) == 3 && hi == 0)
                *(v4f*)(optr[c] + (t - 3)) = outv[c];   // float4 store per row
        }
    }
}

extern "C" void kernel_launch(void* const* d_in, const int* in_sizes, int n_in,
                              void* d_out, int out_size, void* d_ws, size_t ws_size,
                              hipStream_t stream) {
    const float* features = (const float*)d_in[0];
    const float* W1 = (const float*)d_in[1];
    const float* b1 = (const float*)d_in[2];
    const float* W2 = (const float*)d_in[3];
    const float* b2 = (const float*)d_in[4];
    const float* W3 = (const float*)d_in[5];
    const float* b3 = (const float*)d_in[6];
    float* out = (float*)d_out;

    const int batch = in_sizes[0] / (NSTEPS * STATE_DIM);            // 32768
    const int tiles = (batch + 15) / 16;                             // 2048
    const int waves = (tiles + TILES_PER_WAVE - 1) / TILES_PER_WAVE; // 1024
    const int blocks = (waves + WAVES_PER_BLOCK - 1) / WAVES_PER_BLOCK;  // 128

    DDPGHedge_kernel<<<blocks, WAVES_PER_BLOCK * 32, 0, stream>>>(
        features, W1, b1, W2, b2, W3, b3, out, batch);
}
